// BernNet_21801253994545
// MI455X (gfx1250) — compile-verified, matrix operand order checked by reading, not run on hardware
//
#include <hip/hip_runtime.h>
#include <hip/hip_bf16.h>

// ---------------------------------------------------------------------------
// BernNet on MI455X (gfx1250, wave32).
//  * MLP fused into one kernel using v_wmma_f32_16x16x32_bf16 (16x16x32 tiles)
//  * Bernstein polynomial re-expanded into monomial basis -> only K=10 SpMMs
//  * SpMM: gather (float4) + global_atomic_add_f32 scatter
// ---------------------------------------------------------------------------

typedef __bf16 bf16;
typedef __attribute__((ext_vector_type(16))) __bf16 v16bf;
typedef __attribute__((ext_vector_type(8)))  float  v8f;

#define N_NODES 100000
#define N_EDGES 3200000
#define NFEAT   512
#define HIDDEN  128
#define NCLASS  64
#define KORD    10

// ---------------------------------------------------------------------------
// Monomial coefficients: c_j = (1/2^K) * sum_k relu(temp_k)*C(K,k)*e[k][j]
// e[k][j] = coefficient of A^j in (1+A)^(K-k) (1-A)^k   (K=10)
// ---------------------------------------------------------------------------
__global__ void coeff_kernel(const float* __restrict__ temp,
                             float* __restrict__ c,
                             float* __restrict__ out_tail) {
  if (blockIdx.x != 0 || threadIdx.x != 0) return;
  const float binom[11] = {1,10,45,120,210,252,210,120,45,10,1};
  const float e[11][11] = {
    {1, 10, 45, 120, 210, 252, 210, 120, 45, 10, 1},
    {1,  8, 27,  48,  42,   0, -42, -48,-27, -8,-1},
    {1,  6, 13,   8, -14, -28, -14,   8, 13,  6, 1},
    {1,  4,  3,  -8, -14,   0,  14,   8, -3, -4,-1},
    {1,  2, -3,  -8,   2,  12,   2,  -8, -3,  2, 1},
    {1,  0, -5,   0,  10,   0, -10,   0,  5,  0,-1},
    {1, -2, -3,   8,   2, -12,   2,   8, -3, -2, 1},
    {1, -4,  3,   8, -14,   0,  14,  -8, -3,  4,-1},
    {1, -6, 13,  -8, -14,  28, -14,  -8, 13, -6, 1},
    {1, -8, 27, -48,  42,   0, -42,  48,-27,  8,-1},
    {1,-10, 45,-120, 210,-252, 210,-120, 45,-10, 1}};
  float T[11];
  for (int k = 0; k <= KORD; ++k) {
    float t = temp[k];
    T[k] = t > 0.f ? t : 0.f;
    out_tail[k] = T[k];             // second tuple output: TEMP
  }
  for (int j = 0; j <= KORD; ++j) {
    float s = 0.f;
    for (int k = 0; k <= KORD; ++k) s += T[k] * binom[k] * e[k][j];
    c[j] = s * (1.0f / 1024.0f);
  }
}

// ---------------------------------------------------------------------------
// Pack W1/W2 (fp32 row-major [K][N]) into wave32 bf16 B-fragment layout:
// idx = ((kb*NT + ntile)*32 + lane)*16 + j
//   lane: col = ntile*16 + (lane&15), half = lane>>4
//   j<8  -> K = kb*32 + half*8 + j
//   j>=8 -> K = kb*32 + 16 + half*8 + (j-8)
// ---------------------------------------------------------------------------
__global__ void pack_w1_kernel(const float* __restrict__ W, bf16* __restrict__ P) {
  int tid = blockIdx.x * blockDim.x + threadIdx.x;       // 65536
  int j = tid & 15, lane = (tid >> 4) & 31, nt = (tid >> 9) & 7, kb = tid >> 12;
  int half = lane >> 4, n = lane & 15;
  int col = nt * 16 + n;
  int kk = kb * 32 + (j < 8 ? half * 8 + j : 16 + half * 8 + (j - 8));
  P[tid] = (bf16)W[kk * HIDDEN + col];
}

__global__ void pack_w2_kernel(const float* __restrict__ W, bf16* __restrict__ P) {
  int tid = blockIdx.x * blockDim.x + threadIdx.x;       // 8192
  int j = tid & 15, lane = (tid >> 4) & 31, nt = (tid >> 9) & 3, kb = tid >> 11;
  int half = lane >> 4, n = lane & 15;
  int col = nt * 16 + n;
  int kk = kb * 32 + (j < 8 ? half * 8 + j : 16 + half * 8 + (j - 8));
  P[tid] = (bf16)W[kk * NCLASS + col];
}

// ---------------------------------------------------------------------------
// Fused MLP: h = relu(x@W1+b1)@W2 + b2   (one 16-row block per workgroup)
// Stage 1: 8 waves x one 16x16 N-tile each, K=512 (16 WMMAs/wave)
// Stage 2: 4 waves x one 16x16 N-tile each, K=128 (4 WMMAs/wave)
// ---------------------------------------------------------------------------
__global__ __launch_bounds__(256) void mlp_kernel(
    const float* __restrict__ x, const float* __restrict__ b1,
    const float* __restrict__ b2, const bf16* __restrict__ pW1,
    const bf16* __restrict__ pW2, float* __restrict__ h) {
  __shared__ bf16 xs[16][NFEAT + 8];     // +8 bf16 pad -> no LDS bank conflicts
  __shared__ bf16 h1s[16][HIDDEN + 8];

  const int row0 = blockIdx.x * 16;
  const int t = threadIdx.x;

  // Cooperative load + fp32->bf16 convert of the 16x512 x tile into LDS
  for (int i = t; i < 16 * NFEAT; i += 256) {
    int m = i >> 9, kk = i & (NFEAT - 1);
    xs[m][kk] = (bf16)x[(size_t)(row0 + m) * NFEAT + kk];
  }
  __syncthreads();

  const int wave = t >> 5;
  const int lane = t & 31;
  const int m = lane & 15, half = lane >> 4;

  // ---- Stage 1: x @ W1 (per-wave N-tile = wave) ----
  v8f acc = {};
  for (int kb = 0; kb < NFEAT / 32; ++kb) {
    v16bf a, b;
    const bf16* pa0 = &xs[m][kb * 32 + half * 8];
    const bf16* pa1 = pa0 + 16;
#pragma unroll
    for (int i = 0; i < 8; ++i) { a[i] = pa0[i]; a[8 + i] = pa1[i]; }
    b = *(const v16bf*)(pW1 + (((size_t)kb * 8 + wave) * 32 + lane) * 16);
    acc = __builtin_amdgcn_wmma_f32_16x16x32_bf16(false, a, false, b,
                                                  (short)0, acc, false, false);
  }
  // bias + relu -> bf16 h1 tile in LDS (C layout: vgpr r, lanes<16 -> M=r)
  {
    const int mb = (lane >= 16) ? 8 : 0;
    const int col = wave * 16 + (lane & 15);
    const float bias = b1[col];
#pragma unroll
    for (int r = 0; r < 8; ++r) {
      float v = acc[r] + bias;
      h1s[mb + r][col] = (bf16)(v > 0.f ? v : 0.f);
    }
  }
  __syncthreads();

  // ---- Stage 2: h1 @ W2 (waves 0..3, N-tile = wave) ----
  if (wave < 4) {
    v8f acc2 = {};
    for (int kb = 0; kb < HIDDEN / 32; ++kb) {
      v16bf a, b;
      const bf16* pa0 = &h1s[m][kb * 32 + half * 8];
      const bf16* pa1 = pa0 + 16;
#pragma unroll
      for (int i = 0; i < 8; ++i) { a[i] = pa0[i]; a[8 + i] = pa1[i]; }
      b = *(const v16bf*)(pW2 + (((size_t)kb * 4 + wave) * 32 + lane) * 16);
      acc2 = __builtin_amdgcn_wmma_f32_16x16x32_bf16(false, a, false, b,
                                                     (short)0, acc2, false, false);
    }
    const int mb = (lane >= 16) ? 8 : 0;
    const int col = wave * 16 + (lane & 15);
    const float bias = b2[col];
#pragma unroll
    for (int r = 0; r < 8; ++r)
      h[(size_t)(row0 + mb + r) * NCLASS + col] = acc2[r] + bias;
  }
}

// ---------------------------------------------------------------------------
// Graph pieces
// ---------------------------------------------------------------------------
__global__ void zero_f32_kernel(float* __restrict__ p, long n) {
  long i = (long)blockIdx.x * blockDim.x + threadIdx.x;
  for (; i < n; i += (long)gridDim.x * blockDim.x) p[i] = 0.f;
}

__global__ void deg_kernel(const int* __restrict__ src, float* __restrict__ deg) {
  int e = blockIdx.x * blockDim.x + threadIdx.x;
  if (e < N_EDGES) atomicAdd(&deg[src[e]], 1.0f);
}

__global__ void dinv_kernel(const float* __restrict__ deg, float* __restrict__ dinv) {
  int i = blockIdx.x * blockDim.x + threadIdx.x;
  if (i < N_NODES) {
    float d = deg[i];
    dinv[i] = d > 0.f ? rsqrtf(d) : 0.f;
  }
}

__global__ void edgew_kernel(const int* __restrict__ ei,
                             const float* __restrict__ dinv,
                             float* __restrict__ w) {
  int e = blockIdx.x * blockDim.x + threadIdx.x;
  if (e < N_EDGES) w[e] = dinv[ei[e]] * dinv[ei[N_EDGES + e]];
}

// vout[dst] += w * vin[src]  : 16 half-lanes per edge, float4 per thread.
// Lanes 0..15 of a wave read one contiguous 256B row of vin -> coalesced.
__global__ void spmv_kernel(const int* __restrict__ ei, const float* __restrict__ w,
                            const float* __restrict__ vin, float* __restrict__ vout) {
  unsigned tid = blockIdx.x * blockDim.x + threadIdx.x;   // N_EDGES*16 threads
  int e = tid >> 4;
  int f0 = (tid & 15) * 4;
  int s = ei[e], d = ei[N_EDGES + e];
  float wv = w[e];
  const float4 v = *(const float4*)(vin + (size_t)s * NCLASS + f0);
  float* o = vout + (size_t)d * NCLASS + f0;
  atomicAdd(o + 0, wv * v.x);
  atomicAdd(o + 1, wv * v.y);
  atomicAdd(o + 2, wv * v.z);
  atomicAdd(o + 3, wv * v.w);
}

// acc = (init ? 0 : acc) + c[j] * v
__global__ void axpy_kernel(float* __restrict__ acc, const float* __restrict__ v,
                            const float* __restrict__ c, int j, int init) {
  long i = (long)blockIdx.x * blockDim.x + threadIdx.x;
  const long n = (long)N_NODES * NCLASS;
  const float cv = c[j];
  for (; i < n; i += (long)gridDim.x * blockDim.x)
    acc[i] = (init ? 0.f : acc[i]) + cv * v[i];
}

// row-wise log_softmax, one wave32 per row (2 features per lane)
__global__ void lsm_kernel(const float* __restrict__ acc, float* __restrict__ out) {
  int row = blockIdx.x * 8 + (threadIdx.x >> 5);
  int lane = threadIdx.x & 31;
  if (row >= N_NODES) return;
  const float* p = acc + (size_t)row * NCLASS;
  float a = p[lane], b = p[lane + 32];
  float m = fmaxf(a, b);
#pragma unroll
  for (int off = 16; off; off >>= 1) m = fmaxf(m, __shfl_xor(m, off, 32));
  float s = __expf(a - m) + __expf(b - m);
#pragma unroll
  for (int off = 16; off; off >>= 1) s += __shfl_xor(s, off, 32);
  float ls = __logf(s) + m;
  out[(size_t)row * NCLASS + lane] = a - ls;
  out[(size_t)row * NCLASS + lane + 32] = b - ls;
}

// ---------------------------------------------------------------------------
extern "C" void kernel_launch(void* const* d_in, const int* in_sizes, int n_in,
                              void* d_out, int out_size, void* d_ws, size_t ws_size,
                              hipStream_t stream) {
  const float* x    = (const float*)d_in[0];
  const int*   ei   = (const int*)d_in[1];     // [2, E] flat: src then dst
  const float* W1   = (const float*)d_in[2];
  const float* b1   = (const float*)d_in[3];
  const float* W2   = (const float*)d_in[4];
  const float* b2   = (const float*)d_in[5];
  const float* temp = (const float*)d_in[6];
  float* out = (float*)d_out;

  // Workspace carve-out (256B aligned)
  char* w = (char*)d_ws;
  size_t off = 0;
  auto carve = [&](size_t bytes) { void* p = w + off; off += (bytes + 255) & ~(size_t)255; return p; };
  float* c    = (float*)carve(11 * sizeof(float));
  bf16*  pW1  = (bf16*) carve((size_t)NFEAT * HIDDEN * sizeof(bf16));   // 128 KB
  bf16*  pW2  = (bf16*) carve((size_t)HIDDEN * NCLASS * sizeof(bf16));  // 16 KB
  float* deg  = (float*)carve((size_t)N_NODES * sizeof(float));
  float* dinv = (float*)carve((size_t)N_NODES * sizeof(float));
  float* wE   = (float*)carve((size_t)N_EDGES * sizeof(float));         // 12.8 MB
  float* h    = (float*)carve((size_t)N_NODES * NCLASS * sizeof(float)); // 25.6 MB
  float* vbuf = (float*)carve((size_t)N_NODES * NCLASS * sizeof(float)); // 25.6 MB
  float* acc  = (float*)carve((size_t)N_NODES * NCLASS * sizeof(float)); // 25.6 MB
  (void)ws_size; (void)n_in; (void)in_sizes; (void)out_size;

  // Coefficients + TEMP tail output
  coeff_kernel<<<1, 64, 0, stream>>>(temp, c, out + (size_t)N_NODES * NCLASS);

  // Pack weights into WMMA B-fragment layout
  pack_w1_kernel<<<(NFEAT * HIDDEN) / 256, 256, 0, stream>>>(W1, pW1);
  pack_w2_kernel<<<(HIDDEN * NCLASS) / 256, 256, 0, stream>>>(W2, pW2);

  // Fused MLP -> h
  mlp_kernel<<<N_NODES / 16, 256, 0, stream>>>(x, b1, b2, pW1, pW2, h);

  // Degrees, inverse sqrt, edge weights
  zero_f32_kernel<<<(N_NODES + 255) / 256, 256, 0, stream>>>(deg, N_NODES);
  deg_kernel<<<N_EDGES / 256, 256, 0, stream>>>(ei, deg);
  dinv_kernel<<<(N_NODES + 255) / 256, 256, 0, stream>>>(deg, dinv);
  edgew_kernel<<<N_EDGES / 256, 256, 0, stream>>>(ei, dinv, wE);

  // acc = c0*h ; then 10x (vnext = A vcur ; acc += c_j*vnext)
  const long nv = (long)N_NODES * NCLASS;
  axpy_kernel<<<25000, 256, 0, stream>>>(acc, h, c, 0, 1);
  float* vcur = h;
  float* vnext = vbuf;
  for (int j = 1; j <= KORD; ++j) {
    zero_f32_kernel<<<(int)((nv + 255) / 256), 256, 0, stream>>>(vnext, nv);
    spmv_kernel<<<(N_EDGES * 16) / 256, 256, 0, stream>>>(ei, wE, vcur, vnext);
    axpy_kernel<<<25000, 256, 0, stream>>>(acc, vnext, c, j, 0);
    float* t = vcur; vcur = vnext; vnext = t;
  }

  // log_softmax -> d_out
  lsm_kernel<<<N_NODES / 8, 256, 0, stream>>>(acc, out);
}
